// ExplicitDecoderJoint_39779987095672
// MI455X (gfx1250) — matrix operand order repaired and checked by reading, not compiled, not run on hardware
//
#include <hip/hip_runtime.h>
#include <hip/hip_bf16.h>
#include <math.h>

typedef __attribute__((ext_vector_type(16))) _Float16 v16h;
typedef __attribute__((ext_vector_type(8)))  float    v8f;
typedef __attribute__((ext_vector_type(8)))  _Float16 h8;
typedef __attribute__((ext_vector_type(4)))  float    f4;

#define B_   32
#define T_   512
#define DE_  512
#define H_   640
#define V_   4096
#define M_   (B_*T_)    /* 16384 */
#define G4_  (4*H_)     /* 2560 */

// ---------------------------------------------------------------------------
// small helpers
// ---------------------------------------------------------------------------
__device__ __forceinline__ float sig_(float x) { return 1.0f / (1.0f + expf(-x)); }

// ---------------------------------------------------------------------------
// fp32 -> fp16 weight conversion
// ---------------------------------------------------------------------------
__global__ void cvt_f16_kernel(const float* __restrict__ src,
                               _Float16* __restrict__ dst, int n) {
  int i = blockIdx.x * blockDim.x + threadIdx.x;
  if (i < n) dst[i] = (_Float16)src[i];
}

// ---------------------------------------------------------------------------
// LSTM gate GEMV: gates[b][q] = x[b]·w_ih[q] + b_ih[q] + h[b]·w_hh[q] + b_hh[q]
// layer 0 uses embedding lookup for x (tok != nullptr)
// ---------------------------------------------------------------------------
__global__ void lstm_gates_kernel(const float* __restrict__ x,
                                  const long long* __restrict__ tok,
                                  const float* __restrict__ embed,
                                  const float* __restrict__ hprev,
                                  const float* __restrict__ w_ih,
                                  const float* __restrict__ w_hh,
                                  const float* __restrict__ b_ih,
                                  const float* __restrict__ b_hh,
                                  float* __restrict__ gates) {
  int id = blockIdx.x * blockDim.x + threadIdx.x;
  if (id >= B_ * G4_) return;
  int b = id / G4_;
  int q = id % G4_;
  const float* xr = (tok != nullptr) ? (embed + (size_t)tok[b] * H_)
                                     : (x + (size_t)b * H_);
  const float* hr  = hprev + (size_t)b * H_;
  const float* wi  = w_ih + (size_t)q * H_;
  const float* wh  = w_hh + (size_t)q * H_;
  float acc = b_ih[q] + b_hh[q];
  #pragma unroll 4
  for (int k = 0; k < H_; k += 4) {
    f4 xv = *(const f4*)(xr + k);
    f4 wv = *(const f4*)(wi + k);
    f4 hv = *(const f4*)(hr + k);
    f4 gv = *(const f4*)(wh + k);
    acc += xv.x * wv.x + xv.y * wv.y + xv.z * wv.z + xv.w * wv.w;
    acc += hv.x * gv.x + hv.y * gv.y + hv.z * gv.z + hv.w * gv.w;
  }
  gates[(size_t)b * G4_ + q] = acc;
}

// ---------------------------------------------------------------------------
// LSTM activations: writes h,c into workspace and into d_out state slices
// ---------------------------------------------------------------------------
__global__ void lstm_act_kernel(const float* __restrict__ gates,
                                const float* __restrict__ cprev,
                                float* __restrict__ h_ws, float* __restrict__ c_ws,
                                float* __restrict__ h_out, float* __restrict__ c_out) {
  int id = blockIdx.x * blockDim.x + threadIdx.x;
  if (id >= B_ * H_) return;
  int b = id / H_;
  int j = id % H_;
  const float* g = gates + (size_t)b * G4_;
  float ig = sig_(g[j]);
  float fg = sig_(g[H_ + j]);
  float gg = tanhf(g[2 * H_ + j]);
  float og = sig_(g[3 * H_ + j]);
  float cn = fg * cprev[(size_t)b * H_ + j] + ig * gg;
  float hn = og * tanhf(cn);
  h_ws[id] = hn;  c_ws[id] = cn;
  h_out[id] = hn; c_out[id] = cn;
}

// ---------------------------------------------------------------------------
// gfull[b][j] = h1[b]·Wp[j] + bp[j] + be[j]   (folds both joint biases)
// ---------------------------------------------------------------------------
__global__ void joint_pred_kernel(const float* __restrict__ h1,
                                  const float* __restrict__ wp,
                                  const float* __restrict__ bp,
                                  const float* __restrict__ be,
                                  float* __restrict__ gfull) {
  int id = blockIdx.x * blockDim.x + threadIdx.x;
  if (id >= B_ * H_) return;
  int b = id / H_;
  int j = id % H_;
  const float* hr = h1 + (size_t)b * H_;
  const float* wr = wp + (size_t)j * H_;
  float acc = bp[j] + be[j];
  #pragma unroll 4
  for (int k = 0; k < H_; k += 4) {
    f4 hv = *(const f4*)(hr + k);
    f4 wv = *(const f4*)(wr + k);
    acc += hv.x * wv.x + hv.y * wv.y + hv.z * wv.z + hv.w * wv.w;
  }
  gfull[id] = acc;
}

__global__ void copy_len_kernel(const int* __restrict__ src, int* __restrict__ dst) {
  int i = threadIdx.x;
  if (i < B_) dst[i] = src[i];
}

// ---------------------------------------------------------------------------
// WMMA fragment helpers (ISA 7.12.2 layouts, wave32)
// ---------------------------------------------------------------------------
__device__ __forceinline__ v16h load_a_frag(const _Float16* At, int mrow, int hi) {
  // A 16x32 f16: lanes 0-15 (row M): K {0..7,16..23}; lanes 16-31: K {8..15,24..31}
  int bk = hi ? 8 : 0;
  h8 lo = *reinterpret_cast<const h8*>(&At[mrow * 40 + bk]);
  h8 hi8 = *reinterpret_cast<const h8*>(&At[mrow * 40 + bk + 16]);
  v16h a;
  #pragma unroll
  for (int i = 0; i < 8; ++i) { a[i] = lo[i]; a[8 + i] = hi8[i]; }
  return a;
}

__device__ __forceinline__ v16h load_b_frag(const _Float16* Bt, int nrow, int hi) {
  // B 32x16 f16 (stored [n][k] in LDS): lanes 0-15 K {0..7,16..23}; lanes 16-31: K {8..15,24..31}
  int kb = hi ? 8 : 0;
  h8 lo = *reinterpret_cast<const h8*>(&Bt[nrow * 40 + kb]);
  h8 hi8 = *reinterpret_cast<const h8*>(&Bt[nrow * 40 + kb + 16]);
  v16h b;
  #pragma unroll
  for (int i = 0; i < 8; ++i) { b[i] = lo[i]; b[8 + i] = hi8[i]; }
  return b;
}

// ---------------------------------------------------------------------------
// GEMM1: ajoint[m][n] = f16( relu( sum_d enc[b][d][t]*We[n][d] + gfull[b][n] ) )
// M=16384, K=512, N=640.  grid (5, 128), 256 threads (8 waves)
// Software-pipelined: global loads for K-step k+1 in flight during WMMAs of k;
// explicit two-deep B-fragment rotation so each ds_load has a WMMA of cover.
// ---------------------------------------------------------------------------
__global__ __launch_bounds__(256) void gemm1_kernel(
    const float* __restrict__ enc,       // (B, DE, T)
    const _Float16* __restrict__ we16,   // (640, 512)
    const float* __restrict__ gfull,     // (B, 640)
    _Float16* __restrict__ ajoint) {     // (M, 640)
  __shared__ _Float16 At[128 * 40];
  __shared__ _Float16 Bt[128 * 40];
  const int n0 = blockIdx.x * 128;
  const int m0 = blockIdx.y * 128;
  const int b = m0 >> 9;         // 512 rows per batch, tile never crosses batch
  const int tbase = m0 & 511;
  const int tid = threadIdx.x;
  const int wave = tid >> 5;
  const int lane = tid & 31;
  const int lane15 = lane & 15;
  const int hi = lane >> 4;

  // A staging: thread owns k-pair kp (k = 2*kp, 2*kp+1) and 8 m's
  const int kp  = tid >> 4;          // 0..15
  const int mc8 = (tid & 15) << 3;   // 0..120
  const float* arow = enc + ((size_t)(b * DE_) + 2 * kp) * T_ + tbase + mc8;
  // B staging: thread owns n row (tid>>1) and half of k (ko)
  const int nrw = tid >> 1;
  const int ko  = (tid & 1) << 4;
  const _Float16* brow = we16 + (size_t)(n0 + nrw) * 512 + ko;

  v8f acc[8] = {};

  // prologue: fetch K-step 0 into registers
  f4 e00 = *(const f4*)(arow);
  f4 e01 = *(const f4*)(arow + 4);
  f4 e10 = *(const f4*)(arow + T_);
  f4 e11 = *(const f4*)(arow + T_ + 4);
  h8 bs0 = *(const h8*)(brow);
  h8 bs1 = *(const h8*)(brow + 8);

  for (int k0 = 0; k0 < 512; k0 += 32) {
    // commit staged tile to LDS: pack (k even, k odd) into one dword store
    #pragma unroll
    for (int i = 0; i < 8; ++i) {
      float lo = (i < 4) ? e00[i] : e01[i - 4];
      float hf = (i < 4) ? e10[i] : e11[i - 4];
      union { _Float16 h[2]; unsigned u; } p;
      p.h[0] = (_Float16)lo; p.h[1] = (_Float16)hf;
      *reinterpret_cast<unsigned*>(&At[(mc8 + i) * 40 + 2 * kp]) = p.u;
    }
    *reinterpret_cast<h8*>(&Bt[nrw * 40 + ko])     = bs0;
    *reinterpret_cast<h8*>(&Bt[nrw * 40 + ko + 8]) = bs1;
    __syncthreads();

    // issue next K-step's global loads (overlap with WMMA below)
    if (k0 + 32 < 512) {
      const float* an = arow + (size_t)(k0 + 32) * T_;
      e00 = *(const f4*)(an);
      e01 = *(const f4*)(an + 4);
      e10 = *(const f4*)(an + T_);
      e11 = *(const f4*)(an + T_ + 4);
      bs0 = *(const h8*)(brow + k0 + 32);
      bs1 = *(const h8*)(brow + k0 + 40);
    }

    // compute: explicit two-deep B-fragment rotation
    v16h a = load_a_frag(At, wave * 16 + lane15, hi);
    v16h bcur = load_b_frag(Bt, lane15, hi);
    #pragma unroll
    for (int t = 0; t < 7; ++t) {
      v16h bnxt = load_b_frag(Bt, (t + 1) * 16 + lane15, hi);
      acc[t] = __builtin_amdgcn_wmma_f32_16x16x32_f16(
          false, a, false, bcur, (short)0, acc[t], false, false);
      bcur = bnxt;
    }
    acc[7] = __builtin_amdgcn_wmma_f32_16x16x32_f16(
        false, a, false, bcur, (short)0, acc[7], false, false);
    __syncthreads();
  }

  const int mbase = m0 + wave * 16 + hi * 8;
  #pragma unroll
  for (int t = 0; t < 8; ++t) {
    const int n = n0 + t * 16 + lane15;
    const float gv = gfull[(size_t)b * H_ + n];
    #pragma unroll
    for (int r = 0; r < 8; ++r) {
      float v = acc[t][r] + gv;
      v = v > 0.0f ? v : 0.0f;
      ajoint[(size_t)(mbase + r) * H_ + n] = (_Float16)v;
    }
  }
}

// ---------------------------------------------------------------------------
// GEMM2: out[m][v] = sum_j ajoint[m][j]*Wout[v][j] + bias[v]
// M=16384, K=640, N=4096.  grid (32, 64), 256 threads (8 waves)
// Block tile 256(M)x128(N): each wave owns TWO 16-row A fragments sharing all
// B fragments -> 2 WMMAs per B-fragment LDS load; explicit two-deep B rotation.
// ---------------------------------------------------------------------------
__global__ __launch_bounds__(256) void gemm2_kernel(
    const _Float16* __restrict__ ajoint,  // (M, 640)
    const _Float16* __restrict__ wout16,  // (4096, 640)
    const float* __restrict__ bias,       // (4096)
    float* __restrict__ out) {            // (M, 4096)
  __shared__ _Float16 At[256 * 40];
  __shared__ _Float16 Bt[128 * 40];
  const int n0 = blockIdx.x * 128;
  const int m0 = blockIdx.y * 256;
  const int tid = threadIdx.x;
  const int wave = tid >> 5;
  const int lane = tid & 31;
  const int lane15 = lane & 15;
  const int hi = lane >> 4;

  // A staging: one of 256 tile rows per thread (32 halves)
  const _Float16* aptr = ajoint + (size_t)(m0 + tid) * H_;
  // B staging: thread owns n row (tid>>1) and half of k
  const int nrw = tid >> 1;
  const int ko  = (tid & 1) << 4;
  const _Float16* bptr = wout16 + (size_t)(n0 + nrw) * H_ + ko;

  v8f acc0[8] = {};
  v8f acc1[8] = {};

  // prologue: fetch K-step 0
  h8 as0 = *(const h8*)(aptr);
  h8 as1 = *(const h8*)(aptr + 8);
  h8 as2 = *(const h8*)(aptr + 16);
  h8 as3 = *(const h8*)(aptr + 24);
  h8 bs0 = *(const h8*)(bptr);
  h8 bs1 = *(const h8*)(bptr + 8);

  for (int k0 = 0; k0 < H_; k0 += 32) {
    *reinterpret_cast<h8*>(&At[tid * 40 +  0]) = as0;
    *reinterpret_cast<h8*>(&At[tid * 40 +  8]) = as1;
    *reinterpret_cast<h8*>(&At[tid * 40 + 16]) = as2;
    *reinterpret_cast<h8*>(&At[tid * 40 + 24]) = as3;
    *reinterpret_cast<h8*>(&Bt[nrw * 40 + ko])     = bs0;
    *reinterpret_cast<h8*>(&Bt[nrw * 40 + ko + 8]) = bs1;
    __syncthreads();

    // issue next K-step's global loads (overlap with WMMA below)
    if (k0 + 32 < H_) {
      as0 = *(const h8*)(aptr + k0 + 32);
      as1 = *(const h8*)(aptr + k0 + 40);
      as2 = *(const h8*)(aptr + k0 + 48);
      as3 = *(const h8*)(aptr + k0 + 56);
      bs0 = *(const h8*)(bptr + k0 + 32);
      bs1 = *(const h8*)(bptr + k0 + 40);
    }

    v16h a0 = load_a_frag(At, wave * 32 + lane15, hi);
    v16h a1 = load_a_frag(At, wave * 32 + 16 + lane15, hi);
    v16h bcur = load_b_frag(Bt, lane15, hi);
    #pragma unroll
    for (int t = 0; t < 7; ++t) {
      v16h bnxt = load_b_frag(Bt, (t + 1) * 16 + lane15, hi);
      acc0[t] = __builtin_amdgcn_wmma_f32_16x16x32_f16(
          false, a0, false, bcur, (short)0, acc0[t], false, false);
      acc1[t] = __builtin_amdgcn_wmma_f32_16x16x32_f16(
          false, a1, false, bcur, (short)0, acc1[t], false, false);
      bcur = bnxt;
    }
    acc0[7] = __builtin_amdgcn_wmma_f32_16x16x32_f16(
        false, a0, false, bcur, (short)0, acc0[7], false, false);
    acc1[7] = __builtin_amdgcn_wmma_f32_16x16x32_f16(
        false, a1, false, bcur, (short)0, acc1[7], false, false);
    __syncthreads();
  }

  const int mbase0 = m0 + wave * 32 + hi * 8;
  const int mbase1 = mbase0 + 16;
  #pragma unroll
  for (int t = 0; t < 8; ++t) {
    const int n = n0 + t * 16 + lane15;
    const float bv = bias[n];
    #pragma unroll
    for (int r = 0; r < 8; ++r) {
      out[(size_t)(mbase0 + r) * V_ + n] = acc0[t][r] + bv;
      out[(size_t)(mbase1 + r) * V_ + n] = acc1[t][r] + bv;
    }
  }
}

// ---------------------------------------------------------------------------
// launch
// ---------------------------------------------------------------------------
extern "C" void kernel_launch(void* const* d_in, const int* in_sizes, int n_in,
                              void* d_out, int out_size, void* d_ws, size_t ws_size,
                              hipStream_t stream) {
  const float*     enc      = (const float*)d_in[0];
  const long long* targets  = (const long long*)d_in[1];
  const int*       tgt_len  = (const int*)d_in[2];
  const float*     states1  = (const float*)d_in[3];   // (2,B,H)
  const float*     states2  = (const float*)d_in[4];
  const float*     w_ih0    = (const float*)d_in[5];
  const float*     w_hh0    = (const float*)d_in[6];
  const float*     b_ih0    = (const float*)d_in[7];
  const float*     b_hh0    = (const float*)d_in[8];
  const float*     w_ih1    = (const float*)d_in[9];
  const float*     w_hh1    = (const float*)d_in[10];
  const float*     b_ih1    = (const float*)d_in[11];
  const float*     b_hh1    = (const float*)d_in[12];
  const float*     embed    = (const float*)d_in[13];
  const float*     we       = (const float*)d_in[14];  // (640,512)
  const float*     be       = (const float*)d_in[15];
  const float*     wp       = (const float*)d_in[16];  // (640,640)
  const float*     bp       = (const float*)d_in[17];
  const float*     wout     = (const float*)d_in[18];  // (4096,640)
  const float*     bout     = (const float*)d_in[19];

  // workspace layout (all offsets 256B aligned)
  char* ws = (char*)d_ws;
  _Float16* ajoint = (_Float16*)(ws);                    // 16384*640*2  = 20,971,520
  _Float16* wout16 = (_Float16*)(ws + 20971520);         // 4096*640*2   =  5,242,880
  _Float16* we16   = (_Float16*)(ws + 26214400);         // 640*512*2    =    655,360
  float* gates0    = (float*)(ws + 26869760);            // 32*2560*4
  float* gates1    = (float*)(ws + 27197440);
  float* h0        = (float*)(ws + 27525120);            // 32*640*4 each
  float* c0        = (float*)(ws + 27607040);
  float* h1        = (float*)(ws + 27688960);
  float* c1        = (float*)(ws + 27770880);
  float* gfull     = (float*)(ws + 27852800);

  // d_out layout: outputs | target_length | states1 | states2
  float* out     = (float*)d_out;
  int*   len_out = (int*)((float*)d_out + (size_t)M_ * V_);            // 67,108,864
  float* s1_out  = (float*)d_out + (size_t)M_ * V_ + B_;               // (2,B,H)
  float* s2_out  = s1_out + 2 * B_ * H_;

  // weight conversions (independent of LSTM chain)
  cvt_f16_kernel<<<(V_ * H_ + 255) / 256, 256, 0, stream>>>(wout, wout16, V_ * H_);
  cvt_f16_kernel<<<(H_ * DE_ + 255) / 256, 256, 0, stream>>>(we, we16, H_ * DE_);

  // LSTM layer 0
  lstm_gates_kernel<<<(B_ * G4_ + 255) / 256, 256, 0, stream>>>(
      nullptr, targets, embed, states1 /*h0 prev*/, w_ih0, w_hh0, b_ih0, b_hh0, gates0);
  lstm_act_kernel<<<(B_ * H_ + 255) / 256, 256, 0, stream>>>(
      gates0, states2 /*c0 prev*/, h0, c0, s1_out, s2_out);

  // LSTM layer 1
  lstm_gates_kernel<<<(B_ * G4_ + 255) / 256, 256, 0, stream>>>(
      h0, nullptr, nullptr, states1 + B_ * H_, w_ih1, w_hh1, b_ih1, b_hh1, gates1);
  lstm_act_kernel<<<(B_ * H_ + 255) / 256, 256, 0, stream>>>(
      gates1, states2 + B_ * H_, h1, c1, s1_out + B_ * H_, s2_out + B_ * H_);

  // g vector (folds joint_pred_bias + joint_enc_bias)
  joint_pred_kernel<<<(B_ * H_ + 255) / 256, 256, 0, stream>>>(h1, wp, bp, be, gfull);

  // GEMM1: joint = relu(enc^T @ We^T + gfull) -> f16
  gemm1_kernel<<<dim3(H_ / 128, M_ / 128), 256, 0, stream>>>(enc, we16, gfull, ajoint);

  // GEMM2: outputs = joint @ Wout^T + bias  (256x128 tiles)
  gemm2_kernel<<<dim3(V_ / 128, M_ / 256), 256, 0, stream>>>(ajoint, wout16, bout, out);

  // pass-through target_length
  copy_len_kernel<<<1, 32, 0, stream>>>(tgt_len, len_out);
}
